// MultiQueryAttention_28312424415851
// MI455X (gfx1250) — matrix operand order repaired and checked by reading, not compiled, not run on hardware
//
#include <hip/hip_runtime.h>

#define BB 2
#define TT 2048
#define DD 2048
#define HH 16
#define HD 128

typedef __attribute__((ext_vector_type(16))) __bf16 v16bf;
typedef __attribute__((ext_vector_type(8)))  float  v8f;
typedef __attribute__((ext_vector_type(16))) unsigned short v16u;
typedef __attribute__((ext_vector_type(8)))  unsigned short v8u;

__device__ __forceinline__ unsigned short f2bf(float x) {
  unsigned u = __float_as_uint(x);
  u += 0x7FFFu + ((u >> 16) & 1u);            // round-to-nearest-even
  return (unsigned short)(u >> 16);
}

__device__ __forceinline__ v8f zero8() {
  v8f z;
#pragma unroll
  for (int i = 0; i < 8; ++i) z[i] = 0.f;
  return z;
}

__device__ __forceinline__ v8f wmma_bf16(v16u a, v16u b, v8f c) {
  return __builtin_amdgcn_wmma_f32_16x16x32_bf16(
      false, __builtin_bit_cast(v16bf, a),
      false, __builtin_bit_cast(v16bf, b),
      (short)0, c, false, false);
}

// Async 16-byte global -> LDS copy (gfx1250, ASYNCcnt-tracked).
__device__ __forceinline__ void async_b128(unsigned ldsOff, const unsigned short* g) {
  asm volatile("global_load_async_to_lds_b128 %0, %1, off"
               :: "v"(ldsOff), "v"(g) : "memory");
}
#define WAIT_ASYNC(n) asm volatile("s_wait_asynccnt %0" :: "n"(n) : "memory")

__device__ __forceinline__ unsigned lds_off(const void* p) {
  return (unsigned)(size_t)p;                 // LDS aperture: addr[31:0] = LDS byte offset
}

// A fragment: 16x32 bf16 tile, row-major source, leading dim ld (elements).
// lanes 0-15: M=lane, K chunks {0..7, 16..23}; lanes 16-31: same M, K {8..15, 24..31}.
__device__ __forceinline__ v16u load_fragA(const unsigned short* p, int ld, int lane) {
  int r  = lane & 15;
  int ko = (lane >> 4) << 3;                  // 0 or 8
  const unsigned short* base = p + (size_t)r * ld + ko;
  v8u c0 = *(const v8u*)(base);
  v8u c1 = *(const v8u*)(base + 16);
  v16u f;
#pragma unroll
  for (int i = 0; i < 8; ++i) { f[i] = c0[i]; f[8 + i] = c1[i]; }
  return f;
}

// B fragment: 32x16 bf16; column n in lane (n, n+16). Source is row-major N x K
// (each source row is one B column). lanes 0-15: K {0..7, 16..23}; lanes 16-31: K {8..15, 24..31}.
__device__ __forceinline__ v16u load_fragB(const unsigned short* p, int ld, int lane) {
  int r  = lane & 15;
  int ko = (lane >> 4) << 3;                  // 0 or 8
  const unsigned short* base = p + (size_t)r * ld + ko;
  v8u c0 = *(const v8u*)(base);
  v8u c1 = *(const v8u*)(base + 16);
  v16u f;
#pragma unroll
  for (int i = 0; i < 8; ++i) { f[i] = c0[i]; f[8 + i] = c1[i]; }
  return f;
}

// ---------------- elementwise conversion kernels ----------------

__global__ void cvt_f32_bf16_kernel(const float* __restrict__ in,
                                    unsigned short* __restrict__ out, size_t n) {
  size_t i = (size_t)blockIdx.x * blockDim.x + threadIdx.x;
  if (i < n) out[i] = f2bf(in[i]);
}

// in: K x N row-major f32  ->  out: N x K row-major bf16
__global__ void transpose_cvt_kernel(const float* __restrict__ in,
                                     unsigned short* __restrict__ out, int K, int N) {
  size_t i = (size_t)blockIdx.x * blockDim.x + threadIdx.x;
  if (i >= (size_t)K * N) return;
  int n = (int)(i % N);
  int k = (int)(i / N);
  out[(size_t)n * K + k] = f2bf(in[i]);
}

__global__ void rotary_q_kernel(const float* __restrict__ q,
                                unsigned short* __restrict__ qb) {
  int idx = blockIdx.x * blockDim.x + threadIdx.x;  // B*T*H*64 threads
  if (idx >= BB * TT * HH * (HD / 2)) return;
  int d   = idx & 63;
  int h   = (idx >> 6) & (HH - 1);
  int row = idx >> 10;
  int t   = row & (TT - 1);
  float ang = (float)t * __powf(10000.f, -(float)d / 64.f);
  float sn, cs;
  __sincosf(ang, &sn, &cs);
  size_t base = (size_t)row * DD + (size_t)h * HD;
  float x1 = q[base + d], x2 = q[base + 64 + d];
  qb[base + d]      = f2bf(x1 * cs - x2 * sn);
  qb[base + 64 + d] = f2bf(x2 * cs + x1 * sn);
}

__global__ void rotary_k_kernel(const float* __restrict__ k,
                                unsigned short* __restrict__ kb) {
  int idx = blockIdx.x * blockDim.x + threadIdx.x;  // B*T*64 threads
  if (idx >= BB * TT * (HD / 2)) return;
  int d   = idx & 63;
  int row = idx >> 6;
  int t   = row & (TT - 1);
  float ang = (float)t * __powf(10000.f, -(float)d / 64.f);
  float sn, cs;
  __sincosf(ang, &sn, &cs);
  size_t base = (size_t)row * HD;
  float x1 = k[base + d], x2 = k[base + 64 + d];
  kb[base + d]      = f2bf(x1 * cs - x2 * sn);
  kb[base + 64 + d] = f2bf(x2 * cs + x1 * sn);
}

// v: (B*T) x HD f32  ->  vt: B x HD x T bf16
__global__ void transpose_v_kernel(const float* __restrict__ v,
                                   unsigned short* __restrict__ vt) {
  int idx = blockIdx.x * blockDim.x + threadIdx.x;  // B*T*HD threads
  if (idx >= BB * TT * HD) return;
  int dd = idx & (HD - 1);
  int t  = (idx >> 7) & (TT - 1);
  int b  = idx >> 18;
  vt[(size_t)b * HD * TT + (size_t)dd * TT + t] = f2bf(v[idx]);
}

// ---------------- WMMA GEMM: C = A (MxK) * Bt^T (Bt is NxK) + bias ----------------
// 256 threads = 8 waves; wave w owns rows [m0, m0+32) x cols [n0, n0+128)
// => block tile 256 x 128. B tile (128 x 32) async-staged in LDS, double-buffered.
// Steady-state loop is straight-line: issue-next, wait(2), barrier, 16 WMMA, barrier.
__global__ __launch_bounds__(256)
void gemm_bf16_kernel(const unsigned short* __restrict__ A,
                      const unsigned short* __restrict__ Bt,
                      const float* __restrict__ bias,
                      float* __restrict__ C, int M, int N, int K) {
  __shared__ __align__(16) unsigned short Bs[2][128 * 32];
  int tid  = threadIdx.x;
  int lane = tid & 31;
  int wave = tid >> 5;
  int m0 = blockIdx.x * 256 + wave * 32;
  int n0 = blockIdx.y * 128;

  v8f acc[2][8];
#pragma unroll
  for (int r = 0; r < 2; ++r)
#pragma unroll
    for (int i = 0; i < 8; ++i) acc[r][i] = zero8();

  auto issueB = [&](int k, int s) {
#pragma unroll
    for (int c = 0; c < 2; ++c) {
      int ci   = tid * 2 + c;                 // 512 16-byte chunks
      int row  = ci >> 2;                     // 0..127 (B column index)
      int part = (ci & 3) * 8;                // u16 offset inside 32-elem row
      async_b128(lds_off(&Bs[s][row * 32 + part]),
                 Bt + (size_t)(n0 + row) * K + k + part);
    }
  };

  auto compute = [&](int cur, v16u a0, v16u a1) {
    v16u bfr[8];
#pragma unroll
    for (int nb = 0; nb < 8; ++nb)
      bfr[nb] = load_fragB(&Bs[cur][(nb * 16) * 32], 32, lane);
#pragma unroll
    for (int nb = 0; nb < 8; ++nb) acc[0][nb] = wmma_bf16(a0, bfr[nb], acc[0][nb]);
#pragma unroll
    for (int nb = 0; nb < 8; ++nb) acc[1][nb] = wmma_bf16(a1, bfr[nb], acc[1][nb]);
  };

  issueB(0, 0);
  v16u a_cur0 = load_fragA(A + (size_t)m0 * K, K, lane);
  v16u a_cur1 = load_fragA(A + (size_t)(m0 + 16) * K, K, lane);

  const int NI = K / 32;
  for (int it = 0; it < NI - 1; ++it) {
    int cur = it & 1;
    issueB((it + 1) * 32, cur ^ 1);
    WAIT_ASYNC(2);                            // current tile landed; next in flight
    __syncthreads();
    v16u a_nxt0 = load_fragA(A + (size_t)m0 * K + (it + 1) * 32, K, lane);
    v16u a_nxt1 = load_fragA(A + (size_t)(m0 + 16) * K + (it + 1) * 32, K, lane);
    compute(cur, a_cur0, a_cur1);
    a_cur0 = a_nxt0;
    a_cur1 = a_nxt1;
    __syncthreads();                          // reads done before next overwrite
  }
  WAIT_ASYNC(0);
  __syncthreads();
  compute((NI - 1) & 1, a_cur0, a_cur1);

  int col_lo = lane & 15;
  int rhi = (lane >> 4) << 3;
#pragma unroll
  for (int r = 0; r < 2; ++r)
#pragma unroll
    for (int nb = 0; nb < 8; ++nb) {
      int col = n0 + nb * 16 + col_lo;
      float bv = bias[col];
#pragma unroll
      for (int v = 0; v < 8; ++v) {
        int row = m0 + r * 16 + v + rhi;
        C[(size_t)row * N + col] = acc[r][nb][v] + bv;
      }
    }
}

// ---------------- Flash MQA: ctx = softmax(Q K^T / sqrt(HD)) V ----------------
// grid.x = B*H, grid.y = T/128; 8 waves, wave owns 16 q-rows x HD cols.
// K tile (32 tok x 128) and V^T tile (128 hd x 32 tok) shared by all 8 waves:
// async-staged, double-buffered; fragment loads batched; V frags fetched before
// the softmax VALU segment so ds latency hides behind v_exp work.
__global__ __launch_bounds__(256)
void flash_mqa_kernel(const unsigned short* __restrict__ Q,   // (B*T) x D bf16 (rotated)
                      const unsigned short* __restrict__ Kc,  // (B*T) x HD bf16 (rotated)
                      const unsigned short* __restrict__ Vt,  // B x HD x T bf16
                      unsigned short* __restrict__ Ctx) {     // (B*T) x D bf16
  __shared__ __align__(16) unsigned short Ks[2][32 * 128];
  __shared__ __align__(16) unsigned short Vs[2][128 * 32];
  __shared__ __align__(16) unsigned short Pb[8 * 16 * 32];
  int tid  = threadIdx.x;
  int lane = tid & 31;
  int wave = tid >> 5;
  int b = blockIdx.x / HH;
  int h = blockIdx.x % HH;
  int q0 = blockIdx.y * 128 + wave * 16;
  const unsigned short* qp = Q + (size_t)(b * TT + q0) * DD + (size_t)h * HD;
  const unsigned short* kp = Kc + (size_t)b * TT * HD;
  const unsigned short* vp = Vt + (size_t)b * HD * TT;
  unsigned short* Pw = Pb + wave * (16 * 32);

  auto issueKV = [&](int kt, int s) {
#pragma unroll
    for (int c = 0; c < 2; ++c) {
      int ci = tid * 2 + c;                   // K tile: 512 chunks, rows = tokens
      int row  = ci >> 4;                     // 0..31
      int part = (ci & 15) * 8;               // 0..120
      async_b128(lds_off(&Ks[s][row * 128 + part]),
                 kp + (size_t)(kt + row) * HD + part);
    }
#pragma unroll
    for (int c = 0; c < 2; ++c) {
      int ci = tid * 2 + c;                   // V tile: 512 chunks, rows = head-dims
      int row  = ci >> 2;                     // 0..127
      int part = (ci & 3) * 8;                // 0..24
      async_b128(lds_off(&Vs[s][row * 32 + part]),
                 vp + (size_t)row * TT + kt + part);
    }
  };

  v16u qf[4];
#pragma unroll
  for (int kk = 0; kk < 4; ++kk) qf[kk] = load_fragA(qp + kk * 32, DD, lane);

  v8f acc[8];
  float mr[8], lr[8];
#pragma unroll
  for (int i = 0; i < 8; ++i) { acc[i] = zero8(); mr[i] = -1e30f; lr[i] = 0.f; }
  const float sc = 0.08838834764831845f;      // 1/sqrt(128)

  int prow_hi = (lane >> 4) << 3;
  int pcol = lane & 15;

  auto body = [&](int cur) {
    // S = Q * K^T for 32 key tokens: batch all 8 K-fragments, then 8 WMMAs
    v16u kf[8];
#pragma unroll
    for (int kk = 0; kk < 4; ++kk) {
      kf[kk]     = load_fragB(&Ks[cur][kk * 32], 128, lane);
      kf[4 + kk] = load_fragB(&Ks[cur][16 * 128 + kk * 32], 128, lane);
    }
    v8f s0 = zero8(), s1 = zero8();
#pragma unroll
    for (int kk = 0; kk < 4; ++kk) s0 = wmma_bf16(qf[kk], kf[kk], s0);
#pragma unroll
    for (int kk = 0; kk < 4; ++kk) s1 = wmma_bf16(qf[kk], kf[4 + kk], s1);

    // prefetch V fragments; ds latency overlaps the softmax VALU below
    v16u vf[8];
#pragma unroll
    for (int nb = 0; nb < 8; ++nb)
      vf[nb] = load_fragB(&Vs[cur][(nb * 16) * 32], 32, lane);

    // online softmax per row (rows striped: VGPR v -> row v + 8*(lane>=16))
#pragma unroll
    for (int v = 0; v < 8; ++v) {
      float a0 = s0[v] * sc, a1 = s1[v] * sc;
      float mx = fmaxf(a0, a1);
#pragma unroll
      for (int d = 1; d < 16; d <<= 1) mx = fmaxf(mx, __shfl_xor(mx, d, 32));
      float mnew = fmaxf(mr[v], mx);
      float corr = __expf(mr[v] - mnew);
      float p0 = __expf(a0 - mnew);
      float p1 = __expf(a1 - mnew);
      float rs = p0 + p1;
#pragma unroll
      for (int d = 1; d < 16; d <<= 1) rs += __shfl_xor(rs, d, 32);
      lr[v] = lr[v] * corr + rs;
      mr[v] = mnew;
#pragma unroll
      for (int nb = 0; nb < 8; ++nb) acc[nb][v] *= corr;
      int pr = v + prow_hi;
      Pw[pr * 32 + pcol]      = f2bf(p0);
      Pw[pr * 32 + pcol + 16] = f2bf(p1);
    }
    // ctx += P (16x32) * V-tile
    v16u pa = load_fragA(Pw, 32, lane);
#pragma unroll
    for (int nb = 0; nb < 8; ++nb) acc[nb] = wmma_bf16(pa, vf[nb], acc[nb]);
  };

  issueKV(0, 0);
  const int NI = TT / 32;
  for (int it = 0; it < NI - 1; ++it) {
    int cur = it & 1;
    issueKV((it + 1) * 32, cur ^ 1);
    WAIT_ASYNC(4);                            // current K+V tiles landed
    __syncthreads();
    body(cur);
    __syncthreads();                          // reads done before next overwrite
  }
  WAIT_ASYNC(0);
  __syncthreads();
  body((NI - 1) & 1);

  // normalize + store bf16 ctx in (B*T, D) layout
#pragma unroll
  for (int v = 0; v < 8; ++v) {
    float inv = 1.f / lr[v];
    size_t row = (size_t)(b * TT + q0 + v + prow_hi);
#pragma unroll
    for (int nb = 0; nb < 8; ++nb) {
      Ctx[row * DD + (size_t)h * HD + nb * 16 + pcol] = f2bf(acc[nb][v] * inv);
    }
  }
}

// ---------------- launch ----------------

extern "C" void kernel_launch(void* const* d_in, const int* in_sizes, int n_in,
                              void* d_out, int out_size, void* d_ws, size_t ws_size,
                              hipStream_t stream) {
  const float* x  = (const float*)d_in[0];
  const float* Wq = (const float*)d_in[1];
  const float* bq = (const float*)d_in[2];
  const float* Wk = (const float*)d_in[3];
  const float* bk = (const float*)d_in[4];
  const float* Wv = (const float*)d_in[5];
  const float* bv = (const float*)d_in[6];
  const float* Wo = (const float*)d_in[7];
  const float* bo = (const float*)d_in[8];
  float* out = (float*)d_out;

  const size_t M = (size_t)BB * TT;          // 4096 rows
  char* w = (char*)d_ws;
  size_t off = 0;
  auto take = [&](size_t bytes) { char* p = w + off; off += (bytes + 255) & ~(size_t)255; return p; };

  unsigned short* xb   = (unsigned short*)take(M * DD * 2);
  unsigned short* wqT  = (unsigned short*)take((size_t)DD * DD * 2);
  unsigned short* wkT  = (unsigned short*)take((size_t)HD * DD * 2);
  unsigned short* wvT  = (unsigned short*)take((size_t)HD * DD * 2);
  unsigned short* woT  = (unsigned short*)take((size_t)DD * DD * 2);
  float*          qf   = (float*)take(M * DD * 4);
  float*          kf   = (float*)take(M * HD * 4);
  float*          vf   = (float*)take(M * HD * 4);
  unsigned short* qb2  = (unsigned short*)take(M * DD * 2);
  unsigned short* kb2  = (unsigned short*)take(M * HD * 2);
  unsigned short* vtb  = (unsigned short*)take(M * HD * 2);
  unsigned short* ctxb = (unsigned short*)take(M * DD * 2);
  (void)ws_size; (void)in_sizes; (void)n_in; (void)out_size;

  // 1) cast x to bf16; transpose+cast weights
  {
    size_t n = M * DD;
    cvt_f32_bf16_kernel<<<(unsigned)((n + 255) / 256), 256, 0, stream>>>(x, xb, n);
  }
  transpose_cvt_kernel<<<(unsigned)(((size_t)DD * DD + 255) / 256), 256, 0, stream>>>(Wq, wqT, DD, DD);
  transpose_cvt_kernel<<<(unsigned)(((size_t)DD * HD + 255) / 256), 256, 0, stream>>>(Wk, wkT, DD, HD);
  transpose_cvt_kernel<<<(unsigned)(((size_t)DD * HD + 255) / 256), 256, 0, stream>>>(Wv, wvT, DD, HD);
  transpose_cvt_kernel<<<(unsigned)(((size_t)DD * DD + 255) / 256), 256, 0, stream>>>(Wo, woT, DD, DD);

  // 2) projections (WMMA + async LDS staging); block tile 256x128
  gemm_bf16_kernel<<<dim3(16, 16), 256, 0, stream>>>(xb, wqT, bq, qf, (int)M, DD, DD);
  gemm_bf16_kernel<<<dim3(16, 1),  256, 0, stream>>>(xb, wkT, bk, kf, (int)M, HD, DD);
  gemm_bf16_kernel<<<dim3(16, 1),  256, 0, stream>>>(xb, wvT, bv, vf, (int)M, HD, DD);

  // 3) rotary + layout conversion to bf16
  rotary_q_kernel<<<(BB * TT * HH * (HD / 2)) / 256, 256, 0, stream>>>(qf, qb2);
  rotary_k_kernel<<<(BB * TT * (HD / 2)) / 256, 256, 0, stream>>>(kf, kb2);
  transpose_v_kernel<<<(BB * TT * HD) / 256, 256, 0, stream>>>(vf, vtb);

  // 4) fused flash attention (WMMA, K/V async-staged, scores never leave the WGP)
  flash_mqa_kernel<<<dim3(BB * HH, TT / 128), 256, 0, stream>>>(qb2, kb2, vtb, ctxb);

  // 5) output projection (WMMA), fp32 result
  gemm_bf16_kernel<<<dim3(16, 16), 256, 0, stream>>>(ctxb, woT, bo, out, (int)M, DD, DD);
}
